// ViT5MSDeformAttnEncoderLayer_23948737642741
// MI455X (gfx1250) — matrix operand order repaired and checked by reading, not compiled, not run on hardware
//
#include <hip/hip_runtime.h>

// ---------------------------------------------------------------------------
// ViT MSDeformAttn encoder layer for MI455X (gfx1250, wave32, WMMA).
// Memory-bound (~0.5GB traffic @ 23.3TB/s >> 72 GFLOP of GEMM), so:
//  - all GEMMs via v_wmma_f32_16x16x32_f16 (f32 accumulate),
//  - f16 intermediates, fused epilogues (bias/LayerScale/residual/SiLU-mul).
// ---------------------------------------------------------------------------

typedef _Float16 f16;
typedef __attribute__((ext_vector_type(16))) _Float16 v16h;
typedef __attribute__((ext_vector_type(8)))  _Float16 v8h;
typedef __attribute__((ext_vector_type(4)))  _Float16 v4h;
typedef __attribute__((ext_vector_type(8)))  float    v8f;

#define BS   2
#define LQ   17821
#define NTOK (BS * LQ)     // 35642 rows
#define CDIM 256
#define DFFN 1024

// ---------------------------------------------------------------------------
// f32 -> f16 weight cast
// ---------------------------------------------------------------------------
__global__ __launch_bounds__(256) void cvt_f32_f16(const float* __restrict__ in,
                                                   f16* __restrict__ out, int n) {
  for (int i = blockIdx.x * 256 + threadIdx.x; i < n; i += gridDim.x * 256)
    out[i] = (f16)in[i];
}

// ---------------------------------------------------------------------------
// RMSNorm: one wave per row of 256. out1 = f16(norm); out2 = f16(norm + pos).
// ---------------------------------------------------------------------------
__global__ __launch_bounds__(256)
void rmsnorm_kernel(const float* __restrict__ in, const float* __restrict__ pos,
                    const float* __restrict__ w, f16* __restrict__ out1,
                    f16* __restrict__ out2, int nrows) {
  const int wave = threadIdx.x >> 5, lane = threadIdx.x & 31;
  const int row = blockIdx.x * 8 + wave;
  if (row >= nrows) return;  // wave-uniform
  const float4* ip = (const float4*)(in + (size_t)row * CDIM);
  float4 a = ip[lane], b = ip[lane + 32];
  float ss = a.x*a.x + a.y*a.y + a.z*a.z + a.w*a.w
           + b.x*b.x + b.y*b.y + b.z*b.z + b.w*b.w;
#pragma unroll
  for (int m = 1; m <= 16; m <<= 1) ss += __shfl_xor(ss, m, 32);
  const float r = rsqrtf(ss * (1.0f / 256.0f) + 1e-6f);
  const float4* wp = (const float4*)w;
  float4 wa = wp[lane], wb = wp[lane + 32];
  float na0 = a.x*r*wa.x, na1 = a.y*r*wa.y, na2 = a.z*r*wa.z, na3 = a.w*r*wa.w;
  float nb0 = b.x*r*wb.x, nb1 = b.y*r*wb.y, nb2 = b.z*r*wb.z, nb3 = b.w*r*wb.w;
  v4h* o1 = (v4h*)(out1 + (size_t)row * CDIM);
  o1[lane]      = (v4h){(f16)na0, (f16)na1, (f16)na2, (f16)na3};
  o1[lane + 32] = (v4h){(f16)nb0, (f16)nb1, (f16)nb2, (f16)nb3};
  if (out2) {
    const float4* pp = (const float4*)(pos + (size_t)row * CDIM);
    float4 pa = pp[lane], pb = pp[lane + 32];
    v4h* o2 = (v4h*)(out2 + (size_t)row * CDIM);
    o2[lane]      = (v4h){(f16)(na0+pa.x), (f16)(na1+pa.y), (f16)(na2+pa.z), (f16)(na3+pa.w)};
    o2[lane + 32] = (v4h){(f16)(nb0+pb.x), (f16)(nb1+pb.y), (f16)(nb2+pb.z), (f16)(nb3+pb.w)};
  }
}

// ---------------------------------------------------------------------------
// WMMA GEMM: out[n,m] = epi( sum_k A[n,k]*W[m,k] + bias[m] ).
// 8 waves/block; wave = 16x64 tile (4 accumulators). K multiple of 32,
// M multiple of 64. A,W row-major f16.
// EPI: 0 -> f32 out;  1 -> f16 out;  2 -> out = resid + gamma*acc (f32).
// ---------------------------------------------------------------------------
template <int EPI>
__global__ __launch_bounds__(256)
void gemm_kernel(const f16* __restrict__ A, const f16* __restrict__ W,
                 const float* __restrict__ bias, float* __restrict__ outf,
                 f16* __restrict__ outh, const float* __restrict__ resid,
                 const float* __restrict__ gamma, int nrows, int K, int M) {
  const int lane = threadIdx.x & 31;
  const int wave = threadIdx.x >> 5;
  const int l16  = lane & 15;
  const int hi   = lane >> 4;
  const int rowbase = (blockIdx.x * 8 + wave) * 16;
  const int colbase = blockIdx.y * 64;

  int arow = rowbase + l16; if (arow > nrows - 1) arow = nrows - 1;
  const f16* aptr = A + (size_t)arow * K + hi * 16;
  const f16* wptr = W + (size_t)(colbase + l16) * K + hi * 16;

  v8f acc[4];
#pragma unroll
  for (int j = 0; j < 4; ++j) {
    const float bv = bias[colbase + j * 16 + l16];
#pragma unroll
    for (int r = 0; r < 8; ++r) acc[j][r] = bv;
  }

#pragma unroll 4
  for (int k0 = 0; k0 < K; k0 += 32) {
    union { v16h v; v8h h[2]; } a;
    a.h[0] = *(const v8h*)(aptr + k0);
    a.h[1] = *(const v8h*)(aptr + k0 + 8);
#pragma unroll
    for (int j = 0; j < 4; ++j) {
      const v16h b = *(const v16h*)(wptr + (size_t)j * 16 * K + k0);
      acc[j] = __builtin_amdgcn_wmma_f32_16x16x32_f16(
          false, a.v, false, b, (short)0, acc[j], false, false);
    }
  }

#pragma unroll
  for (int j = 0; j < 4; ++j) {
    const int gcol = colbase + j * 16 + l16;
#pragma unroll
    for (int r = 0; r < 8; ++r) {
      const int grow = rowbase + hi * 8 + r;
      if (grow < nrows) {
        const size_t o = (size_t)grow * M + gcol;
        const float v = acc[j][r];
        if constexpr (EPI == 0) outf[o] = v;
        else if constexpr (EPI == 1) outh[o] = (f16)v;
        else outf[o] = resid[o] + gamma[gcol] * v;
      }
    }
  }
}

// ---------------------------------------------------------------------------
// Fused SwiGLU GEMM: h = silu(A*W1^T + b1) * (A*W2^T + b2), f16 out. K=256.
// ---------------------------------------------------------------------------
__global__ __launch_bounds__(256)
void glu_gemm_kernel(const f16* __restrict__ A, const f16* __restrict__ W1,
                     const f16* __restrict__ W2, const float* __restrict__ b1,
                     const float* __restrict__ b2, f16* __restrict__ outh,
                     int nrows, int K, int M) {
  const int lane = threadIdx.x & 31;
  const int wave = threadIdx.x >> 5;
  const int l16  = lane & 15;
  const int hi   = lane >> 4;
  const int rowbase = (blockIdx.x * 8 + wave) * 16;
  const int colbase = blockIdx.y * 64;

  int arow = rowbase + l16; if (arow > nrows - 1) arow = nrows - 1;
  const f16* aptr  = A  + (size_t)arow * K + hi * 16;
  const f16* w1ptr = W1 + (size_t)(colbase + l16) * K + hi * 16;
  const f16* w2ptr = W2 + (size_t)(colbase + l16) * K + hi * 16;

  v8f acc1[4], acc2[4];
#pragma unroll
  for (int j = 0; j < 4; ++j) {
    const float v1 = b1[colbase + j * 16 + l16];
    const float v2 = b2[colbase + j * 16 + l16];
#pragma unroll
    for (int r = 0; r < 8; ++r) { acc1[j][r] = v1; acc2[j][r] = v2; }
  }

#pragma unroll 4
  for (int k0 = 0; k0 < K; k0 += 32) {
    union { v16h v; v8h h[2]; } a;
    a.h[0] = *(const v8h*)(aptr + k0);
    a.h[1] = *(const v8h*)(aptr + k0 + 8);
#pragma unroll
    for (int j = 0; j < 4; ++j) {
      const v16h bb1 = *(const v16h*)(w1ptr + (size_t)j * 16 * K + k0);
      acc1[j] = __builtin_amdgcn_wmma_f32_16x16x32_f16(
          false, a.v, false, bb1, (short)0, acc1[j], false, false);
      const v16h bb2 = *(const v16h*)(w2ptr + (size_t)j * 16 * K + k0);
      acc2[j] = __builtin_amdgcn_wmma_f32_16x16x32_f16(
          false, a.v, false, bb2, (short)0, acc2[j], false, false);
    }
  }

#pragma unroll
  for (int j = 0; j < 4; ++j) {
    const int gcol = colbase + j * 16 + l16;
#pragma unroll
    for (int r = 0; r < 8; ++r) {
      const int grow = rowbase + hi * 8 + r;
      if (grow < nrows) {
        const float x1 = acc1[j][r];
        const float s  = x1 / (1.0f + __expf(-x1));  // silu
        outh[(size_t)grow * M + gcol] = (f16)(s * acc2[j][r]);
      }
    }
  }
}

// ---------------------------------------------------------------------------
// Deformable sampling. Block = token (b,qi); wave = head; lanes = 32 channels.
// Lanes 0-15 (mirrored 16-31) own the 16 sampling points: softmax + location;
// accumulation broadcasts each point via shfl, gathers f16 value rows.
// ---------------------------------------------------------------------------
__global__ __launch_bounds__(256)
void msda_kernel(const float* __restrict__ off, const float* __restrict__ awl,
                 const f16* __restrict__ value, const float* __restrict__ refp,
                 f16* __restrict__ samp) {
  constexpr int HT[4] = {100, 50, 25, 13};
  constexpr int WT[4] = {134, 67, 34, 17};
  constexpr int ST[4] = {0, 13400, 16750, 17600};

  const int tok  = blockIdx.x;           // 0 .. NTOK-1
  const int b    = tok / LQ;
  const int h    = threadIdx.x >> 5;     // head = wave
  const int lane = threadIdx.x & 31;
  const int p16  = lane & 15;            // point id (mirrored in upper half)

  // softmax over the head's 16 logits
  float e = awl[(size_t)tok * 128 + h * 16 + p16];
  float mx = e;
#pragma unroll
  for (int m = 1; m <= 8; m <<= 1) mx = fmaxf(mx, __shfl_xor(mx, m, 32));
  const float ex = __expf(e - mx);
  float sm = ex;
#pragma unroll
  for (int m = 1; m <= 8; m <<= 1) sm += __shfl_xor(sm, m, 32);
  const float awp = ex / sm;

  // sampling location for this lane's point
  const int   lv   = p16 >> 2;
  const float offx = off[(size_t)tok * 256 + h * 32 + p16 * 2 + 0];
  const float offy = off[(size_t)tok * 256 + h * 32 + p16 * 2 + 1];
  const float refx = refp[(size_t)tok * 8 + lv * 2 + 0];
  const float refy = refp[(size_t)tok * 8 + lv * 2 + 1];
  const float Wf = (float)WT[lv], Hf = (float)HT[lv];
  const float sx = (refx + offx / Wf) * Wf - 0.5f;
  const float sy = (refy + offy / Hf) * Hf - 0.5f;

  float acc = 0.0f;
  const size_t vbase = ((size_t)b * LQ) * 256 + h * 32 + lane;
#pragma unroll
  for (int p = 0; p < 16; ++p) {
    const int l  = p >> 2;
    const int Hl = HT[l], Wl = WT[l], St = ST[l];
    const float xx = __shfl(sx, p, 32);
    const float yy = __shfl(sy, p, 32);
    const float wb = __shfl(awp, p, 32);
    const float x0f = floorf(xx), y0f = floorf(yy);
    const float lx = xx - x0f, ly = yy - y0f;
    const int x0 = (int)x0f, y0 = (int)y0f;
#pragma unroll
    for (int c = 0; c < 4; ++c) {
      const int dx = c & 1, dy = c >> 1;
      const int xi = x0 + dx, yi = y0 + dy;
      if (xi >= 0 && xi < Wl && yi >= 0 && yi < Hl) {   // wave-uniform
        const float wgt = (dx ? lx : 1.0f - lx) * (dy ? ly : 1.0f - ly) * wb;
        acc += wgt * (float)value[vbase + (size_t)(St + yi * Wl + xi) * 256];
      }
    }
  }
  samp[(size_t)tok * 256 + h * 32 + lane] = (f16)acc;
}

// ---------------------------------------------------------------------------
// Host orchestration
// ---------------------------------------------------------------------------
extern "C" void kernel_launch(void* const* d_in, const int* in_sizes, int n_in,
                              void* d_out, int out_size, void* d_ws, size_t ws_size,
                              hipStream_t stream) {
  (void)in_sizes; (void)n_in; (void)out_size; (void)ws_size;
  const float* query  = (const float*)d_in[0];
  const float* qpos   = (const float*)d_in[1];
  const float* refp   = (const float*)d_in[2];
  const float* nattnw = (const float*)d_in[5];
  const float* Wv     = (const float*)d_in[6];
  const float* bv     = (const float*)d_in[7];
  const float* Woff   = (const float*)d_in[8];
  const float* boff   = (const float*)d_in[9];
  const float* Waw    = (const float*)d_in[10];
  const float* baw    = (const float*)d_in[11];
  const float* Wo     = (const float*)d_in[12];
  const float* bo     = (const float*)d_in[13];
  const float* lsattn = (const float*)d_in[14];
  const float* nffnw  = (const float*)d_in[15];
  const float* W1     = (const float*)d_in[16];
  const float* b1     = (const float*)d_in[17];
  const float* W2     = (const float*)d_in[18];
  const float* b2     = (const float*)d_in[19];
  const float* W3     = (const float*)d_in[20];
  const float* b3     = (const float*)d_in[21];
  const float* lsffn  = (const float*)d_in[22];
  float* out = (float*)d_out;

  const size_t N = NTOK;
  char* ws = (char*)d_ws;
  // workspace layout (bytes); buffers aliased phase-by-phase
  f16*   normed_h = (f16*)(ws);                       // N*256 f16
  f16*   q_h      = (f16*)(ws + N * 512);             // N*256 f16
  f16*   value_h  = (f16*)(ws + N * 1024);            // N*256 f16
  float* offb     = (float*)(ws + N * 1536);          // N*256 f32
  float* awlb     = (float*)(ws + N * 2560);          // N*128 f32
  float* xb       = (float*)(ws + N * 3072);          // N*256 f32
  f16*   wh       = (f16*)(ws + N * 4096);            // f16 weights
  f16* Wv_h   = wh;                  // 65536
  f16* Woff_h = Wv_h + 65536;        // 65536
  f16* Waw_h  = Woff_h + 65536;      // 32768
  f16* Wo_h   = Waw_h + 32768;       // 65536
  f16* W1_h   = Wo_h + 65536;        // 262144
  f16* W2_h   = W1_h + 262144;       // 262144
  f16* W3_h   = W2_h + 262144;       // 262144
  // aliases after producers are dead:
  f16* samp_h = normed_h;            // after value GEMM
  f16* n2_h   = q_h;                 // after off/aw GEMMs
  f16* h_h    = value_h;             // N*1024 f16 spans value+off+awl regions

  // 1) cast weights to f16
  cvt_f32_f16<<<256, 256, 0, stream>>>(Wv,   Wv_h,   65536);
  cvt_f32_f16<<<256, 256, 0, stream>>>(Woff, Woff_h, 65536);
  cvt_f32_f16<<<128, 256, 0, stream>>>(Waw,  Waw_h,  32768);
  cvt_f32_f16<<<256, 256, 0, stream>>>(Wo,   Wo_h,   65536);
  cvt_f32_f16<<<1024, 256, 0, stream>>>(W1,  W1_h,   262144);
  cvt_f32_f16<<<1024, 256, 0, stream>>>(W2,  W2_h,   262144);
  cvt_f32_f16<<<1024, 256, 0, stream>>>(W3,  W3_h,   262144);

  const int rn = (int)N;
  const dim3 blk(256);
  const dim3 gnorm((rn + 7) / 8);
  const dim3 ggemm((rn + 127) / 128, 4);   // M = 256
  const dim3 ggemm128((rn + 127) / 128, 2); // M = 128
  const dim3 gglu((rn + 127) / 128, 16);   // M = 1024

  // 2) pre-attn RMSNorm: normed_h, q_h = normed + pos
  rmsnorm_kernel<<<gnorm, blk, 0, stream>>>(query, qpos, nattnw, normed_h, q_h, rn);

  // 3) value = normed @ Wv^T + bv   (f16 out)
  gemm_kernel<1><<<ggemm, blk, 0, stream>>>(normed_h, Wv_h, bv, nullptr, value_h,
                                            nullptr, nullptr, rn, 256, 256);
  // 4) off = q @ Woff^T + boff      (f32 out)
  gemm_kernel<0><<<ggemm, blk, 0, stream>>>(q_h, Woff_h, boff, offb, nullptr,
                                            nullptr, nullptr, rn, 256, 256);
  // 5) aw logits = q @ Waw^T + baw  (f32 out)
  gemm_kernel<0><<<ggemm128, blk, 0, stream>>>(q_h, Waw_h, baw, awlb, nullptr,
                                               nullptr, nullptr, rn, 256, 128);
  // 6) softmax + deformable bilinear sampling -> samp_h (f16)
  msda_kernel<<<dim3(rn), blk, 0, stream>>>(offb, awlb, value_h, refp, samp_h);

  // 7) x = query + ls_attn * (samp @ Wo^T + bo)
  gemm_kernel<2><<<ggemm, blk, 0, stream>>>(samp_h, Wo_h, bo, xb, nullptr,
                                            query, lsattn, rn, 256, 256);
  // 8) FFN RMSNorm -> n2_h
  rmsnorm_kernel<<<gnorm, blk, 0, stream>>>(xb, nullptr, nffnw, n2_h, nullptr, rn);

  // 9) h = silu(n2@W1^T+b1) * (n2@W2^T+b2)   (f16 out, M=1024)
  glu_gemm_kernel<<<gglu, blk, 0, stream>>>(n2_h, W1_h, W2_h, b1, b2, h_h,
                                            rn, 256, 1024);
  // 10) out = x + ls_ffn * (h @ W3^T + b3)   (K=1024)
  gemm_kernel<2><<<ggemm, blk, 0, stream>>>(h_h, W3_h, b3, out, nullptr,
                                            xb, lsffn, rn, 1024, 256);
}
